// AdjGen_38732015075534
// MI455X (gfx1250) — matrix operand order repaired and checked by reading, not compiled, not run on hardware
//
#include <hip/hip_runtime.h>
#include <hip/hip_bf16.h>

// ---------------------------------------------------------------------------
// MI455X (gfx1250) graph-MHA.  N=4096, F=E=256, H=4, hd=64.
// bf16 WMMA (f32 accum) for all matmuls, flash-attention (no NxN tensor),
// TDM (tensor_load_to_lds) for attention K-chunk staging with LDS padding,
// double-buffered / software-pipelined LDS staging everywhere,
// fused A*sigmoid(.+bd) epilogue on the last GEMM.
// ---------------------------------------------------------------------------

typedef __attribute__((ext_vector_type(16))) __bf16 v16bf;
typedef __attribute__((ext_vector_type(8)))  __bf16 v8bf;
typedef __attribute__((ext_vector_type(8)))  float  v8f;
typedef __attribute__((ext_vector_type(4)))  unsigned u32x4;
typedef __attribute__((ext_vector_type(8)))  unsigned u32x8;

#define NN   4096
#define FF   256
#define EE   256
#define HH   4
#define HD   64

// native f32 -> bf16 (RTE) : v_cvt on gfx1250
__device__ __forceinline__ __bf16 f2bf(float f) { return (__bf16)f; }

// ---- WMMA wrapper: D = A(16x32 bf16) * B(32x16 bf16) + C(16x16 f32)
__device__ __forceinline__ v8f wmma_bf16(v16bf a, v16bf b, v8f c) {
  return __builtin_amdgcn_wmma_f32_16x16x32_bf16(
      false, a, false, b, (short)0, c, false, false);
}

// ---- A-fragment (16x32 MxK, ISA 7.12.2): per lane two contiguous 16B reads.
__device__ __forceinline__ v16bf load_afrag(const __bf16* base, int half) {
  uint4 lo = *(const uint4*)(base + half * 8);
  uint4 hi = *(const uint4*)(base + 16 + half * 8);
  v16bf out;
  __builtin_memcpy(&out, &lo, 16);
  __builtin_memcpy((char*)&out + 16, &hi, 16);
  return out;
}

// ---- B-fragment (32x16 KxN) from LDS tile stored [n][k]: two 16B reads (k half*8, 16+half*8).
__device__ __forceinline__ v16bf load_bfrag(const __bf16* base_nk, int half) {
  const __bf16* p = base_nk + half * 8;
  uint4 lo = *(const uint4*)(p);
  uint4 hi = *(const uint4*)(p + 16);
  v16bf out;
  __builtin_memcpy(&out, &lo, 16);
  __builtin_memcpy((char*)&out + 16, &hi, 16);
  return out;
}

// ===========================================================================
// Tensor Data Mover: 64 rows x 64 bf16 tile, global row stride EE halfs,
// into LDS with 8-half row pad (32 DW data + 4 DW pad = 144B pitch = APITCH).
// D# per CDNA5 ISA.
// ===========================================================================
#define APITCH 72

__device__ __forceinline__ void tdm_load_tile64(const __bf16* gsrc,
                                                __bf16* lds_dst) {
  unsigned long long ga = (unsigned long long)(uintptr_t)gsrc;
  unsigned lds = (unsigned)(uintptr_t)lds_dst;  // LDS aperture: low 32 bits
  u32x4 g0;
  g0[0] = 1u;                                           // count=1, user desc
  g0[1] = lds;                                          // lds_addr
  g0[2] = (unsigned)ga;                                 // global_addr[31:0]
  g0[3] = (unsigned)((ga >> 32) & 0x01FFFFFFu) | (2u << 30);  // [56:32]|type=2
  u32x8 g1;
  g1[0] = (1u << 16)      // data_size = 2 bytes
        | (1u << 20)      // pad_enable
        | (4u << 22)      // pad_interval: every 32 DWORDs
        | (3u << 25);     // pad_amount: 4 DWORDs (8 halfs)
  g1[1] = (64u << 16);    // tensor_dim0 = 64 (low 16)
  g1[2] = (64u << 16);    // tensor_dim0 hi = 0 ; tensor_dim1 = 64 (low 16)
  g1[3] = (64u << 16);    // tensor_dim1 hi = 0 ; tile_dim0 = 64
  g1[4] = 64u;            // tile_dim1 = 64 ; tile_dim2 = 0
  g1[5] = (unsigned)EE;   // tensor_dim0_stride = 256
  g1[6] = 0u;
  g1[7] = 0u;
  asm volatile("tensor_load_to_lds %0, %1" :: "s"(g0), "s"(g1) : "memory");
}

// ===========================================================================
// Pipelined tiled GEMM:  C[M,N] = Amat[M,K] @ W[N,K]^T
// (all B sources are weights stored [out,in] row-major = B^T already)
// 256 thr = 8 waves (2x4), block tile 128x128, K-step 32, wave tile 64x32.
// Double-buffered LDS; global loads for step s+1 issued before WMMAs of s.
// EPI 0: bf16 store. EPI 1: Adj * sigmoid(acc + bias[col]) f32 store.
// ===========================================================================
#define BM 128
#define BK 32
#define PITCH 40   // 32 + 8 halfs pad (80B rows)

template <typename T> struct Regs;
template <> struct Regs<float>  { float4 v[4]; };
template <> struct Regs<__bf16> { uint4  v[2]; };

__device__ __forceinline__ void load_tile(Regs<float>& r, const float* src,
                                          int ld, int row, int col) {
  const float4* p = (const float4*)(src + (size_t)row * ld + col);
#pragma unroll
  for (int i = 0; i < 4; ++i) r.v[i] = p[i];
}
__device__ __forceinline__ void load_tile(Regs<__bf16>& r, const __bf16* src,
                                          int ld, int row, int col) {
  const uint4* p = (const uint4*)(src + (size_t)row * ld + col);
#pragma unroll
  for (int i = 0; i < 2; ++i) r.v[i] = p[i];
}
__device__ __forceinline__ void store_tile(const Regs<float>& r, __bf16* dst) {
  __bf16 tmp[16];
#pragma unroll
  for (int i = 0; i < 4; ++i) {
    tmp[4 * i + 0] = f2bf(r.v[i].x);
    tmp[4 * i + 1] = f2bf(r.v[i].y);
    tmp[4 * i + 2] = f2bf(r.v[i].z);
    tmp[4 * i + 3] = f2bf(r.v[i].w);
  }
  uint4 lo, hi;
  __builtin_memcpy(&lo, tmp, 16);
  __builtin_memcpy(&hi, tmp + 8, 16);
  ((uint4*)dst)[0] = lo;
  ((uint4*)dst)[1] = hi;
}
__device__ __forceinline__ void store_tile(const Regs<__bf16>& r, __bf16* dst) {
  ((uint4*)dst)[0] = r.v[0];
  ((uint4*)dst)[1] = r.v[1];
}

template <typename TA, int EPI>
__global__ __launch_bounds__(256) void gemm_wmma(
    const TA* __restrict__ Amat, int lda,
    const float* __restrict__ W, int ldw,       // [N][K] row-major (= B^T)
    __bf16* __restrict__ Cbf, float* __restrict__ Cf,
    const float* __restrict__ Adj, const float* __restrict__ bias,
    int M, int Nn, int K) {
  __shared__ __bf16 sA[2][BM * PITCH];
  __shared__ __bf16 sB[2][BM * PITCH];

  const int m0   = blockIdx.y * BM;
  const int n0   = blockIdx.x * BM;
  const int t    = threadIdx.x;
  const int w    = t >> 5;
  const int lane = t & 31;
  const int wr   = w >> 2, wc = w & 3;
  const int half = lane >> 4, ln = lane & 15;
  const int sr   = t >> 1;            // staging row 0..127
  const int scb  = (t & 1) << 4;      // staging col 0 / 16

  v8f c[4][2] = {};
  Regs<TA> ra; Regs<float> rb;

  // prologue: stage step 0 into buffer 0
  load_tile(ra, Amat, lda, m0 + sr, scb);
  load_tile(rb, W, ldw, n0 + sr, scb);
  store_tile(ra, &sA[0][sr * PITCH + scb]);
  store_tile(rb, &sB[0][sr * PITCH + scb]);
  __syncthreads();

  const int nsteps = K / BK;
  for (int s = 0; s < nsteps; ++s) {
    const int buf = s & 1;
    const bool nxt = (s + 1) < nsteps;
    if (nxt) {  // issue next-tile global loads before this tile's WMMAs
      load_tile(ra, Amat, lda, m0 + sr, (s + 1) * BK + scb);
      load_tile(rb, W, ldw, n0 + sr, (s + 1) * BK + scb);
      if (s + 2 < nsteps) {  // L2 prefetch two steps out
        __builtin_prefetch(Amat + (size_t)(m0 + sr) * lda + (s + 2) * BK + scb, 0, 1);
        __builtin_prefetch(W + (size_t)(n0 + sr) * ldw + (s + 2) * BK + scb, 0, 1);
      }
    }

    v16bf a[4], b[2];
#pragma unroll
    for (int rt = 0; rt < 4; ++rt)
      a[rt] = load_afrag(&sA[buf][(wr * 64 + rt * 16 + ln) * PITCH], half);
#pragma unroll
    for (int nt = 0; nt < 2; ++nt)
      b[nt] = load_bfrag(&sB[buf][(wc * 32 + nt * 16 + ln) * PITCH], half);
#pragma unroll
    for (int rt = 0; rt < 4; ++rt)
#pragma unroll
      for (int nt = 0; nt < 2; ++nt)
        c[rt][nt] = wmma_bf16(a[rt], b[nt], c[rt][nt]);

    if (nxt) {
      store_tile(ra, &sA[buf ^ 1][sr * PITCH + scb]);
      store_tile(rb, &sB[buf ^ 1][sr * PITCH + scb]);
    }
    __syncthreads();
  }

#pragma unroll
  for (int rt = 0; rt < 4; ++rt)
#pragma unroll
    for (int nt = 0; nt < 2; ++nt) {
      const int col = n0 + wc * 32 + nt * 16 + ln;
#pragma unroll
      for (int r = 0; r < 8; ++r) {
        const int row = m0 + wr * 64 + rt * 16 + half * 8 + r;
        const size_t idx = (size_t)row * Nn + col;
        float v = c[rt][nt][r];
        if constexpr (EPI == 0) {
          Cbf[idx] = f2bf(v);
        } else {
          float x = v + bias[col];
          float s = 1.0f / (1.0f + __expf(-x));
          Cf[idx] = Adj[idx] * s;
        }
      }
    }
}

// ===========================================================================
// Flash attention, per (64-row block, head). 4 waves x 16 query rows.
// K chunks staged by TDM (async DMA, issued one chunk ahead), V chunks
// register-pipelined and stored transposed [feat][key]. Online softmax
// with diagonal mask; P round-trips per-wave LDS to re-layout C->A frag.
// ===========================================================================
__global__ __launch_bounds__(128) void attn_flash(
    const __bf16* __restrict__ qh, const __bf16* __restrict__ kh,
    const __bf16* __restrict__ vh, __bf16* __restrict__ ctx) {
  __shared__ __bf16 sK[2][64 * APITCH];       // [key][feat]  (TDM-filled)
  __shared__ __bf16 sV[2][64 * APITCH];       // [feat][key]  (transposed)
  __shared__ __bf16 sP[4 * 16 * APITCH];      // per-wave P tile

  const int h    = blockIdx.y;
  const int rb   = blockIdx.x;
  const int t    = threadIdx.x;
  const int w    = t >> 5;
  const int lane = t & 31;
  const int half = lane >> 4;
  const int ln   = lane & 15;
  const int q0   = rb * 64 + w * 16;
  const int fb   = h * HD;
  const int vr   = t >> 1;            // V staging: key row 0..63
  const int vcb  = (t & 1) * 32;      // V staging: feature base 0/32
  const float scale = 0.125f;         // 1/sqrt(64)

  // Q fragments resident in registers for the whole kernel
  v16bf qf[2];
#pragma unroll
  for (int kc = 0; kc < 2; ++kc)
    qf[kc] = load_afrag(qh + (size_t)(q0 + ln) * EE + fb + kc * 32, half);

  v8f acc[4] = {};
  float mrun[8], lrun[8];
#pragma unroll
  for (int r = 0; r < 8; ++r) { mrun[r] = -INFINITY; lrun[r] = 0.0f; }

  // prologue: chunk 0 -> buffer 0
  v8bf vv[4];
#pragma unroll
  for (int i = 0; i < 4; ++i)
    vv[i] = ((const v8bf*)(vh + (size_t)vr * EE + fb + vcb))[i];
  if (w == 0) tdm_load_tile64(kh + fb, &sK[0][0]);
#pragma unroll
  for (int f = 0; f < 32; ++f)
    sV[0][(vcb + f) * APITCH + vr] = vv[f >> 3][f & 7];
  if (w == 0) __builtin_amdgcn_s_wait_tensorcnt(0);
  __syncthreads();

  for (int j0 = 0; j0 < NN; j0 += 64) {
    const int buf = (j0 >> 6) & 1;
    const bool nxt = (j0 + 64) < NN;
    if (nxt) {  // prefetch next chunk: async TDM for K, reg loads for V
      if (w == 0)
        tdm_load_tile64(kh + (size_t)(j0 + 64) * EE + fb, &sK[buf ^ 1][0]);
#pragma unroll
      for (int i = 0; i < 4; ++i)
        vv[i] = ((const v8bf*)(vh + (size_t)(j0 + 64 + vr) * EE + fb + vcb))[i];
    }

    // ---- S = Q K^T * scale, diagonal mask
    v8f sf[4];
#pragma unroll
    for (int nt = 0; nt < 4; ++nt) {
      v8f z = {};
      sf[nt] = z;
#pragma unroll
      for (int kc = 0; kc < 2; ++kc) {
        v16bf b = load_bfrag(&sK[buf][(nt * 16 + ln) * APITCH + kc * 32], half);
        sf[nt] = wmma_bf16(qf[kc], b, sf[nt]);
      }
      const int j = j0 + nt * 16 + ln;
#pragma unroll
      for (int r = 0; r < 8; ++r) {
        float v = sf[nt][r] * scale;
        if (q0 + half * 8 + r == j) v = -INFINITY;   // exclude self
        sf[nt][r] = v;
      }
    }

    // ---- online softmax (rows replicated across 16-lane halves)
    float rmax[8];
#pragma unroll
    for (int r = 0; r < 8; ++r)
      rmax[r] = fmaxf(fmaxf(sf[0][r], sf[1][r]), fmaxf(sf[2][r], sf[3][r]));
#pragma unroll
    for (int m = 1; m <= 8; m <<= 1)
#pragma unroll
      for (int r = 0; r < 8; ++r)
        rmax[r] = fmaxf(rmax[r], __shfl_xor(rmax[r], m, 32));

    float alpha[8];
#pragma unroll
    for (int r = 0; r < 8; ++r) {
      float mn = fmaxf(mrun[r], rmax[r]);
      alpha[r] = __expf(mrun[r] - mn);
      mrun[r]  = mn;
    }

    __bf16* pb = sP + w * 16 * APITCH;
    float rsum[8] = {0, 0, 0, 0, 0, 0, 0, 0};
#pragma unroll
    for (int nt = 0; nt < 4; ++nt)
#pragma unroll
      for (int r = 0; r < 8; ++r) {
        float p = __expf(sf[nt][r] - mrun[r]);
        rsum[r] += p;
        pb[(half * 8 + r) * APITCH + nt * 16 + ln] = f2bf(p);
      }
#pragma unroll
    for (int m = 1; m <= 8; m <<= 1)
#pragma unroll
      for (int r = 0; r < 8; ++r)
        rsum[r] += __shfl_xor(rsum[r], m, 32);
#pragma unroll
    for (int r = 0; r < 8; ++r) lrun[r] = lrun[r] * alpha[r] + rsum[r];
#pragma unroll
    for (int ft = 0; ft < 4; ++ft)
#pragma unroll
      for (int r = 0; r < 8; ++r) acc[ft][r] *= alpha[r];

    // ---- ctx += P @ V  (same-wave LDS is in-order -> stores visible)
#pragma unroll
    for (int ft = 0; ft < 4; ++ft)
#pragma unroll
      for (int pc = 0; pc < 2; ++pc) {
        v16bf pa = load_afrag(pb + ln * APITCH + pc * 32, half);
        v16bf vb = load_bfrag(&sV[buf][(ft * 16 + ln) * APITCH + pc * 32], half);
        acc[ft] = wmma_bf16(pa, vb, acc[ft]);
      }

    if (nxt) {  // commit next V chunk (transposed) into the other buffer
#pragma unroll
      for (int f = 0; f < 32; ++f)
        sV[buf ^ 1][(vcb + f) * APITCH + vr] = vv[f >> 3][f & 7];
    }
    if (w == 0) __builtin_amdgcn_s_wait_tensorcnt(0);
    __syncthreads();
  }

  // ---- normalize, store bf16 ctx
#pragma unroll
  for (int ft = 0; ft < 4; ++ft)
#pragma unroll
    for (int r = 0; r < 8; ++r) {
      float v = acc[ft][r] / lrun[r];
      ctx[(size_t)(q0 + half * 8 + r) * EE + fb + ft * 16 + ln] = f2bf(v);
    }
}

// ===========================================================================
extern "C" void kernel_launch(void* const* d_in, const int* in_sizes, int n_in,
                              void* d_out, int out_size, void* d_ws,
                              size_t ws_size, hipStream_t stream) {
  (void)in_sizes; (void)n_in; (void)out_size; (void)ws_size;
  const float* A  = (const float*)d_in[0];
  const float* X  = (const float*)d_in[1];
  const float* Wq = (const float*)d_in[2];
  const float* Wk = (const float*)d_in[3];
  const float* Wv = (const float*)d_in[4];
  const float* Wo = (const float*)d_in[5];
  const float* Wd = (const float*)d_in[6];
  const float* bd = (const float*)d_in[7];
  float* out = (float*)d_out;

  __bf16* qh   = (__bf16*)d_ws;                 // 5 bf16 [4096,256] = 10 MB
  __bf16* kh   = qh + (size_t)NN * EE;
  __bf16* vhp  = kh + (size_t)NN * EE;
  __bf16* ctxh = vhp + (size_t)NN * EE;
  __bf16* outh = ctxh + (size_t)NN * EE;

  const dim3 blk(256);
  const dim3 gSmall(EE / BM, NN / BM);
  const dim3 gBig(NN / BM, NN / BM);

  gemm_wmma<float, 0><<<gSmall, blk, 0, stream>>>(X, FF, Wq, EE, qh, nullptr,
                                                  nullptr, nullptr, NN, EE, FF);
  gemm_wmma<float, 0><<<gSmall, blk, 0, stream>>>(X, FF, Wk, FF, kh, nullptr,
                                                  nullptr, nullptr, NN, EE, FF);
  gemm_wmma<float, 0><<<gSmall, blk, 0, stream>>>(A, NN, Wv, NN, vhp, nullptr,
                                                  nullptr, nullptr, NN, EE, NN);

  attn_flash<<<dim3(NN / 64, HH), dim3(128), 0, stream>>>(qh, kh, vhp, ctxh);

  gemm_wmma<__bf16, 0><<<gSmall, blk, 0, stream>>>(ctxh, EE, Wo, EE, outh,
                                                   nullptr, nullptr, nullptr,
                                                   NN, EE, EE);
  gemm_wmma<__bf16, 1><<<gBig, blk, 0, stream>>>(outh, EE, Wd, EE, nullptr,
                                                 out, A, bd, NN, NN, EE);
}
